// MultiHeadAttention_36112085025201
// MI455X (gfx1250) — matrix operand order repaired and checked by reading, not compiled, not run on hardware
//
#include <hip/hip_runtime.h>
#include <hip/hip_bf16.h>

typedef unsigned int   u32;
typedef unsigned short u16;
typedef __attribute__((ext_vector_type(16))) __bf16 v16bf;
typedef __attribute__((ext_vector_type(8)))  float  v8f;
typedef __attribute__((ext_vector_type(4)))  u32    v4u;

#define BB 2
#define SS 4096
#define DD 512
#define HH 8
#define DH 64
#define MM (BB*SS)      // 8192 rows
#define NT16 (DD/16)    // 32 n-tiles
#define KT32 (DD/32)    // 16 k-steps

#if __has_builtin(__builtin_amdgcn_tensor_load_to_lds)
#define HAVE_TDM 1
#endif

// fragment container: 8 dwords == 16 bf16
union AF { v4u q[2]; v16bf h; };

__device__ __forceinline__ u16 f2bf(float f){
  u32 u = __float_as_uint(f);
  u32 r = u + 0x7FFFu + ((u >> 16) & 1u);  // round-to-nearest-even
  return (u16)(r >> 16);
}

#ifdef HAVE_TDM
typedef __attribute__((ext_vector_type(4))) u32 tdmv4u;
typedef __attribute__((ext_vector_type(8))) int tdmv8i;
typedef __attribute__((ext_vector_type(4))) int tdmv4i;

// 2-D tensor tile -> LDS via the Tensor Data Mover (D# per ISA 08_async_tensor §8).
// data_size = 2 bytes (bf16). tensor: [dim1][dim0] row-major with row stride
// stride0 (elements); tile: [tile1][tile0] at gaddr; packed to LDS at lds_off.
__device__ __forceinline__ void tdm_load_2d(u32 lds_off, const void* gaddr,
                                            u32 tdim0, u32 tdim1,
                                            u32 tile0, u32 tile1, u32 stride0){
  unsigned long long ga = (unsigned long long)(size_t)gaddr;
  tdmv4u g0;
  g0[0] = 1u;                                            // count=1, user descriptor
  g0[1] = lds_off;                                       // D#.lds_addr (bytes)
  g0[2] = (u32)(ga & 0xffffffffu);                       // global_addr[31:0]
  g0[3] = (u32)((ga >> 32) & 0x01ffffffu) | (2u << 30);  // addr[56:32] | type=2
  tdmv8i g1;
  g1[0] = (int)(1u << 16);                               // wg_mask=0, data_size=1 (2B)
  g1[1] = (int)((tdim0 & 0xffffu) << 16);                // tensor_dim0[15:0]
  g1[2] = (int)(((tdim0 >> 16) & 0xffffu) | ((tdim1 & 0xffffu) << 16));
  g1[3] = (int)(((tdim1 >> 16) & 0xffffu) | (tile0 << 16)); // tile_dim0
  g1[4] = (int)(tile1 & 0xffffu);                        // tile_dim1 (tile_dim2=0)
  g1[5] = (int)stride0;                                  // tensor_dim0_stride[31:0]
  g1[6] = 0;                                             // stride hi / dim1_stride
  g1[7] = 0;
  tdmv4i z4 = {0, 0, 0, 0};                              // 2-D: groups 2/3 unused
  tdmv8i z8 = {0, 0, 0, 0, 0, 0, 0, 0};
  __builtin_amdgcn_tensor_load_to_lds(g0, g1, z4, z4, z8, 0);
}
#endif

// ---------------- fp32 -> bf16 elementwise ----------------
__global__ void cvt_bf16_kernel(const float* __restrict__ x, u16* __restrict__ o, int n){
  int i = blockIdx.x * blockDim.x + threadIdx.x;
  if (i < n) o[i] = f2bf(x[i]);
}

// ---------------- pack weight [K=in][N=out] into WMMA-B fragment order ----------
__global__ void pack_w_kernel(const float* __restrict__ w, u32* __restrict__ o){
  int d  = blockIdx.x * blockDim.x + threadIdx.x;   // 16*32*32*8 = 131072 dwords
  int j  = d & 7;
  int ln = (d >> 3) & 31;
  int nt = (d >> 8) & 31;
  int kt = d >> 13;
  int n  = nt * 16 + (ln & 15);
  int k  = kt * 32 + ((j & 4) ? 16 : 0) + (ln >> 4) * 8 + 2 * (j & 3);
  u32 lo = f2bf(w[(k    ) * DD + n]);
  u32 hi = f2bf(w[(k + 1) * DD + n]);
  o[d] = lo | (hi << 16);
}

// ---------------- WMMA GEMM (pipelined; constant-offset fragment streams) ------
// ab: byte pointer to this wave's A row start (row*DD + kq elements)
// bb: byte pointer to this wave's B fragment base (lane + n64 folded in)
// Per-step deltas are compile-time: A +64B/step, B +32KB/step, +1KB/tile.
__device__ __forceinline__ void load_step(const char* __restrict__ ab,
                                          const char* __restrict__ bb,
                                          int kt, AF& fa, AF* fb){
  fa.q[0] = *(const v4u*)(ab + kt * 64);
  fa.q[1] = *(const v4u*)(ab + kt * 64 + 32);
#pragma unroll
  for (int nt = 0; nt < 4; ++nt){
    fb[nt].q[0] = *(const v4u*)(bb + kt * 32768 + nt * 1024);
    fb[nt].q[1] = *(const v4u*)(bb + kt * 32768 + nt * 1024 + 16);
  }
}

// mode 0: bf16 out [B,H,S,Dh] (Q,K); mode 1: bf16 out [B,H,Dh,S] (V);
// mode 2: fp32 out row-major [M,D] (final projection)
__global__ void gemm_wmma_kernel(const u16* __restrict__ A, const u32* __restrict__ Wpk,
                                 const float* __restrict__ bias, void* __restrict__ outp,
                                 int mode){
  int lane = threadIdx.x & 31;
  int wvid = threadIdx.x >> 5;
  int t    = blockIdx.x * (blockDim.x >> 5) + wvid;
  int mt   = t & (MM/16 - 1);
  int n64  = t / (MM/16);
  int lm   = lane & 15;
  int kq   = (lane >> 4) * 8;

  const char* ab = (const char*)A +
                   (((size_t)(mt * 16 + lm)) * DD + (size_t)kq) * 2;
  const char* bb = (const char*)Wpk +
                   (((size_t)(n64 * 128 + lane)) * 8) * 4;   // (n64*4 tiles)*32 lanes

  v8f acc[4] = {};
  AF fa_c, fa_n;
  AF fb_c[4], fb_n[4];

  load_step(ab, bb, 0, fa_c, fb_c);
#pragma unroll
  for (int kt = 0; kt < KT32; ++kt){
    if (kt + 1 < KT32)
      load_step(ab, bb, kt + 1, fa_n, fb_n);   // prefetch next k-step
#pragma unroll
    for (int nt = 0; nt < 4; ++nt)
      acc[nt] = __builtin_amdgcn_wmma_f32_16x16x32_bf16(
          false, fa_c.h, false, fb_c[nt].h, (short)0, acc[nt], false, false);
    fa_c = fa_n;
#pragma unroll
    for (int nt = 0; nt < 4; ++nt) fb_c[nt] = fb_n[nt];
  }

  int mrow0 = mt * 16 + ((lane >> 4) * 8);
#pragma unroll
  for (int nt = 0; nt < 4; ++nt){
    int n0 = n64 * 64 + nt * 16 + lm;
    float bvl = bias[n0];
#pragma unroll
    for (int i = 0; i < 8; ++i){
      float v = acc[nt][i] + bvl;
      int r = mrow0 + i;
      if (mode == 2){
        ((float*)outp)[r * DD + n0] = v;
      } else {
        int b = r >> 12, s = r & (SS - 1);
        int h = n0 >> 6, dh = n0 & (DH - 1);
        u16* ob = (u16*)outp;
        if (mode == 0) ob[((b * HH + h) * SS + s ) * DH + dh] = f2bf(v);
        else           ob[((b * HH + h) * DH + dh) * SS + s ] = f2bf(v);
      }
    }
  }
}

// ---------------- flash attention: TDM stages K/V tiles into LDS ---------------
__global__ void flash_wmma_kernel(const u16* __restrict__ Q, const u16* __restrict__ K,
                                  const u16* __restrict__ Vt, u16* __restrict__ O){
  __shared__ __align__(16) u16 ldsK[32 * 64];       // K tile  [32 keys][64 dh]
  __shared__ __align__(16) u16 ldsV[64 * 32];       // Vt tile [64 dh ][32 keys]
  __shared__ __align__(16) u16 ldsP[4 * 16 * 32];   // per-wave P relayout

  int lane = threadIdx.x & 31;
  int wvid = threadIdx.x >> 5;
  int bh   = blockIdx.y;                   // b*H + h
  int qb   = blockIdx.x * 64 + wvid * 16;  // query row base in S
  int hi16 = lane >> 4;
  int lm   = lane & 15;
  int kq   = hi16 * 16;

  const u16* Qb = Q  + (size_t)bh * SS * DH;
  const u16* Kb = K  + (size_t)bh * SS * DH;
  const u16* Vb = Vt + (size_t)bh * DH * SS;
  u16* myp = ldsP + wvid * (16 * 32);

  // Q fragments for Dh=64 (two k-steps of 32)
  AF fq[2];
#pragma unroll
  for (int kt = 0; kt < 2; ++kt){
    const u16* qp = Qb + (qb + lm) * DH + kt * 32 + kq;
    fq[kt].q[0] = *(const v4u*)(qp);
    fq[kt].q[1] = *(const v4u*)(qp + 8);
  }

  v8f o[4] = {};
  float mrow[8], lrow[8];
#pragma unroll
  for (int i = 0; i < 8; ++i){ mrow[i] = -3.0e38f; lrow[i] = 0.f; }

  const float SC = 0.125f;  // 1/sqrt(Dh)

#ifdef HAVE_TDM
  u32 offK = (u32)(size_t)(void*)ldsK;   // flat->LDS: low 32 bits are the LDS byte addr
  u32 offV = (u32)(size_t)(void*)ldsV;
#endif

  for (int kb = 0; kb < SS; kb += 32){
#ifdef HAVE_TDM
    // wave 0 DMAs this key block's K and V tiles into LDS, then publishes them.
    if (wvid == 0){
      tdm_load_2d(offK, (const void*)(Kb + kb * DH), DH, SS, DH, 32, DH);
      tdm_load_2d(offV, (const void*)(Vb + kb),      SS, DH, 32, DH, SS);
      __builtin_amdgcn_s_wait_tensorcnt(0);
    }
    __syncthreads();
#endif
    // ---- S = Q K^T (two 16x16 tiles: keys kb..kb+15 and kb+16..kb+31) ----
    v8f s0 = {}, s1 = {};
#pragma unroll
    for (int kt = 0; kt < 2; ++kt){
#ifdef HAVE_TDM
      const u16* kp0 = ldsK + (lm     ) * DH + kt * 32 + hi16 * 16;
      const u16* kp1 = ldsK + (16 + lm) * DH + kt * 32 + hi16 * 16;
#else
      const u16* kp0 = Kb + (kb      + lm) * DH + kt * 32 + hi16 * 16;
      const u16* kp1 = Kb + (kb + 16 + lm) * DH + kt * 32 + hi16 * 16;
#endif
      AF fk0, fk1;
      fk0.q[0] = *(const v4u*)(kp0); fk0.q[1] = *(const v4u*)(kp0 + 8);
      fk1.q[0] = *(const v4u*)(kp1); fk1.q[1] = *(const v4u*)(kp1 + 8);
      s0 = __builtin_amdgcn_wmma_f32_16x16x32_bf16(false, fq[kt].h, false, fk0.h, (short)0, s0, false, false);
      s1 = __builtin_amdgcn_wmma_f32_16x16x32_bf16(false, fq[kt].h, false, fk1.h, (short)0, s1, false, false);
    }

    // ---- online softmax (rows live across 16-lane groups in C layout) ----
    float al[8];
#pragma unroll
    for (int i = 0; i < 8; ++i){
      float a0 = s0[i] * SC, a1 = s1[i] * SC;
      float mx = fmaxf(a0, a1);
#pragma unroll
      for (int d = 1; d < 16; d <<= 1) mx = fmaxf(mx, __shfl_xor(mx, d, 32));
      float mn = fmaxf(mrow[i], mx);
      float a  = __expf(mrow[i] - mn);
      float p0 = __expf(a0 - mn);
      float p1 = __expf(a1 - mn);
      float rs = p0 + p1;
#pragma unroll
      for (int d = 1; d < 16; d <<= 1) rs += __shfl_xor(rs, d, 32);
      lrow[i] = lrow[i] * a + rs;
      mrow[i] = mn;
      al[i] = a;
      s0[i] = p0; s1[i] = p1;
    }
#pragma unroll
    for (int nt = 0; nt < 4; ++nt)
#pragma unroll
      for (int i = 0; i < 8; ++i) o[nt][i] *= al[i];

    // ---- P: C-layout registers -> LDS -> A-fragment ----
    int mput = hi16 * 8;
#pragma unroll
    for (int i = 0; i < 8; ++i){
      myp[(mput + i) * 32 + lm     ] = f2bf(s0[i]);
      myp[(mput + i) * 32 + 16 + lm] = f2bf(s1[i]);
    }
    AF fp;
    fp.q[0] = *(const v4u*)(myp + lm * 32 + kq);
    fp.q[1] = *(const v4u*)(myp + lm * 32 + kq + 8);

    // ---- O += P @ V ----
#pragma unroll
    for (int nt = 0; nt < 4; ++nt){
#ifdef HAVE_TDM
      const u16* vp = ldsV + (nt * 16 + lm) * 32 + hi16 * 16;
#else
      const u16* vp = Vb + (nt * 16 + lm) * SS + kb + hi16 * 16;
#endif
      AF fv;
      fv.q[0] = *(const v4u*)(vp);
      fv.q[1] = *(const v4u*)(vp + 8);
      o[nt] = __builtin_amdgcn_wmma_f32_16x16x32_bf16(
          false, fp.h, false, fv.h, (short)0, o[nt], false, false);
    }
#ifdef HAVE_TDM
    __syncthreads();   // all waves done with tiles before wave 0 overwrites them
#endif
  }

  // ---- epilogue: normalize, write bf16 row-major [M, D] for final GEMM ----
  int b = bh >> 3, h = bh & (HH - 1);
  int mrow0 = hi16 * 8;
#pragma unroll
  for (int nt = 0; nt < 4; ++nt){
#pragma unroll
    for (int i = 0; i < 8; ++i){
      int r = b * SS + qb + mrow0 + i;
      int c = h * DH + nt * 16 + lm;
      O[r * DD + c] = f2bf(o[nt][i] / lrow[i]);
    }
  }
}

// ------------------------------- launcher --------------------------------------
extern "C" void kernel_launch(void* const* d_in, const int* in_sizes, int n_in,
                              void* d_out, int out_size, void* d_ws, size_t ws_size,
                              hipStream_t stream) {
  const float* x  = (const float*)d_in[0];
  const float* wq = (const float*)d_in[1];
  const float* bq = (const float*)d_in[2];
  const float* wk = (const float*)d_in[3];
  const float* bk = (const float*)d_in[4];
  const float* wv = (const float*)d_in[5];
  const float* bv = (const float*)d_in[6];
  const float* wo = (const float*)d_in[7];
  const float* bo = (const float*)d_in[8];

  // workspace carve-up (~42 MB total, 16B aligned everywhere)
  char* wsb = (char*)d_ws;
  u16* xbf = (u16*)(wsb + 0);           //  8 MB  x as bf16 [M,D]
  u32* wqp = (u32*)(wsb +  8388608);    // 512 KB packed wq
  u32* wkp = (u32*)(wsb +  8912896);    // 512 KB packed wk
  u32* wvp = (u32*)(wsb +  9437184);    // 512 KB packed wv
  u32* wop = (u32*)(wsb +  9961472);    // 512 KB packed wo
  u16* Qb  = (u16*)(wsb + 10485760);    //  8 MB  Q  [B,H,S,Dh]
  u16* Kb  = (u16*)(wsb + 18874368);    //  8 MB  K  [B,H,S,Dh]
  u16* Vtb = (u16*)(wsb + 27262976);    //  8 MB  V  [B,H,Dh,S]
  u16* Ob  = (u16*)(wsb + 35651584);    //  8 MB  attn out bf16 [M,D]

  int n = MM * DD;                                    // 4,194,304
  cvt_bf16_kernel<<<n / 256, 256, 0, stream>>>(x, xbf, n);
  pack_w_kernel<<<131072 / 256, 256, 0, stream>>>(wq, wqp);
  pack_w_kernel<<<131072 / 256, 256, 0, stream>>>(wk, wkp);
  pack_w_kernel<<<131072 / 256, 256, 0, stream>>>(wv, wvp);
  pack_w_kernel<<<131072 / 256, 256, 0, stream>>>(wo, wop);

  int gblocks = (MM / 16) * (DD / 64) / 8;            // 512 blocks x 8 waves
  gemm_wmma_kernel<<<gblocks, 256, 0, stream>>>(xbf, wqp, bq, Qb,  0);
  gemm_wmma_kernel<<<gblocks, 256, 0, stream>>>(xbf, wkp, bk, Kb,  0);
  gemm_wmma_kernel<<<gblocks, 256, 0, stream>>>(xbf, wvp, bv, Vtb, 1);

  flash_wmma_kernel<<<dim3(SS / 64, BB * HH), 128, 0, stream>>>(Qb, Kb, Vtb, Ob);

  gemm_wmma_kernel<<<gblocks, 256, 0, stream>>>(Ob, wop, bo, d_out, 2);
}
